// GCN_472446402720
// MI455X (gfx1250) — matrix operand order repaired and checked by reading, not compiled, not run on hardware
//
#include <hip/hip_runtime.h>

// GCN forward for MI455X (gfx1250, wave32, WMMA).
// Layer1: tmp1 = (x * ns) @ W1  (WMMA f16->f32, K=128)
//         agg1 = scatter_add(tmp1[src] -> dst)
//         feat2 = relu(agg1*nd + b1) * ns
// Layer2: agg2 = scatter_add(feat2[src] -> dst)
//         out  = (agg2 @ W2) * nd + b2   (WMMA, K=16 zero-padded to 32)
//
// B fragments (W1, W2) are wave-invariant: prepacked once into workspace in
// the per-lane WMMA layout so each wave loads them with 2x b128 per fragment.

typedef __attribute__((ext_vector_type(16))) _Float16 v16h;
typedef __attribute__((ext_vector_type(8)))  float    v8f;

#define IN_F  128
#define HID_F 16
#define OUT_F 64

__global__ void k_zero(float* __restrict__ p, long n) {
  long i = (long)blockIdx.x * blockDim.x + threadIdx.x;
  if (i < n) p[i] = 0.0f;
}

__global__ void k_degree(const int* __restrict__ src, const int* __restrict__ dst,
                         unsigned* __restrict__ deg_src, unsigned* __restrict__ deg_dst,
                         int E) {
  int e = blockIdx.x * blockDim.x + threadIdx.x;
  if (e < E) {
    atomicAdd(&deg_src[src[e]], 1u);
    atomicAdd(&deg_dst[dst[e]], 1u);
  }
}

__global__ void k_norm(const unsigned* __restrict__ deg_src,
                       const unsigned* __restrict__ deg_dst,
                       float* __restrict__ norm_src, float* __restrict__ norm_dst, int N) {
  int n = blockIdx.x * blockDim.x + threadIdx.x;
  if (n < N) {
    norm_src[n] = rsqrtf(fmaxf((float)deg_src[n], 1.0f));
    norm_dst[n] = rsqrtf(fmaxf((float)deg_dst[n], 1.0f));
  }
}

// Prepack B fragments (f16, per-lane WMMA layout [step][lane][16]):
//   W1f: 4 K-steps of W1(128x16):  element e of lane -> W1[(kb + h*8 + e + (e&8))*16 + m]
//   W2f: 4 N-tiles of W2(16x64):   element e of lane -> (e<8) ? W2[(h*8 + e)*64 + t*16 + m] : 0
__global__ void k_prep(const float* __restrict__ W1, const float* __restrict__ W2,
                       _Float16* __restrict__ W1f, _Float16* __restrict__ W2f) {
  int i = blockIdx.x * blockDim.x + threadIdx.x;       // 0..4095
  int j = i & 2047;
  int blk  = j >> 9;                                   // kb-step or N-tile (0..3)
  int lane = (j >> 4) & 31;
  int e    = j & 15;
  int h = lane >> 4, m = lane & 15;
  if (i < 2048) {
    W1f[j] = (_Float16)W1[(size_t)(blk * 32 + h * 8 + e + (e & 8)) * HID_F + m];
  } else if (i < 4096) {
    W2f[j] = (e < 8) ? (_Float16)W2[(size_t)(h * 8 + e) * OUT_F + blk * 16 + m]
                      : (_Float16)0.0f;
  }
}

// One wave per 16-node tile: C(16x16) = A(16x128) * W1(128x16), A = x*norm_src.
__global__ void k_gemm1(const float* __restrict__ x,
                        const v16h* __restrict__ W1f,      // [4][32] fragments
                        const float* __restrict__ norm_src,
                        float* __restrict__ tmp1, int N) {
  int wave = (blockIdx.x * blockDim.x + threadIdx.x) >> 5;
  int lane = threadIdx.x & 31;
  int tiles = (N + 15) >> 4;
  if (wave >= tiles) return;                 // wave-uniform exit, EXEC stays full
  int h = lane >> 4;                         // lane half (0/1)
  int m = lane & 15;                         // row within tile / B column
  int row  = wave * 16 + m;
  int rowc = row < N ? row : N - 1;
  float ns = row < N ? norm_src[rowc] : 0.0f;
  const float* xr = x + (size_t)rowc * IN_F;

  v8f c = {0.f, 0.f, 0.f, 0.f, 0.f, 0.f, 0.f, 0.f};
  #pragma unroll
  for (int s = 0; s < 4; ++s) {
    int kb = s * 32;
    v16h b = W1f[s * 32 + lane];                         // 2x b128, coalesced
    // A runs: K = kb + g*16 + h*8 + w  -> two 8-float (32B-aligned) runs
    float4 r0 = *(const float4*)(xr + kb + h * 8);
    float4 r1 = *(const float4*)(xr + kb + h * 8 + 4);
    float4 r2 = *(const float4*)(xr + kb + 16 + h * 8);
    float4 r3 = *(const float4*)(xr + kb + 16 + h * 8 + 4);
    v16h a;
    a[0]  = (_Float16)(r0.x * ns); a[1]  = (_Float16)(r0.y * ns);
    a[2]  = (_Float16)(r0.z * ns); a[3]  = (_Float16)(r0.w * ns);
    a[4]  = (_Float16)(r1.x * ns); a[5]  = (_Float16)(r1.y * ns);
    a[6]  = (_Float16)(r1.z * ns); a[7]  = (_Float16)(r1.w * ns);
    a[8]  = (_Float16)(r2.x * ns); a[9]  = (_Float16)(r2.y * ns);
    a[10] = (_Float16)(r2.z * ns); a[11] = (_Float16)(r2.w * ns);
    a[12] = (_Float16)(r3.x * ns); a[13] = (_Float16)(r3.y * ns);
    a[14] = (_Float16)(r3.z * ns); a[15] = (_Float16)(r3.w * ns);
    c = __builtin_amdgcn_wmma_f32_16x16x32_f16(false, a, false, b, (short)0, c,
                                               false, false);
  }
  #pragma unroll
  for (int i = 0; i < 8; ++i) {
    int r = wave * 16 + i + h * 8;           // D: row = i + h*8, col = m
    if (r < N) tmp1[(size_t)r * HID_F + m] = c[i];
  }
}

// One thread per (edge, feature): 16 adjacent lanes hit one 64B line of agg[dst].
__global__ void k_scatter(const int* __restrict__ src, const int* __restrict__ dst,
                          const float* __restrict__ feat, float* __restrict__ agg,
                          long EF) {
  long i = (long)blockIdx.x * blockDim.x + threadIdx.x;
  if (i >= EF) return;
  int e = (int)(i >> 4);
  int f = (int)(i & 15);
  int s = src[e], d = dst[e];
  atomicAdd(&agg[(size_t)d * HID_F + f], feat[(size_t)s * HID_F + f]);
}

// h = relu(agg1*norm_dst + b1); feat2 = h*norm_src  (fold next layer's src norm)
__global__ void k_post1(const float* __restrict__ agg1,
                        const float* __restrict__ norm_dst,
                        const float* __restrict__ norm_src,
                        const float* __restrict__ b1,
                        float* __restrict__ feat2, long NF) {
  long i = (long)blockIdx.x * blockDim.x + threadIdx.x;
  if (i >= NF) return;
  int n = (int)(i >> 4);
  int f = (int)(i & 15);
  float v = agg1[i] * norm_dst[n] + b1[f];
  v = fmaxf(v, 0.0f);
  feat2[i] = v * norm_src[n];
}

// One wave per 16-node tile: out(16x64) = agg2(16x16) @ W2(16x64), *norm_dst + b2.
__global__ void k_gemm2(const float* __restrict__ agg2,
                        const v16h* __restrict__ W2f,      // [4][32] fragments
                        const float* __restrict__ norm_dst, const float* __restrict__ b2,
                        float* __restrict__ out, int N) {
  int wave = (blockIdx.x * blockDim.x + threadIdx.x) >> 5;
  int lane = threadIdx.x & 31;
  int tiles = (N + 15) >> 4;
  if (wave >= tiles) return;
  int h = lane >> 4;
  int m = lane & 15;
  int row  = wave * 16 + m;
  int rowc = row < N ? row : N - 1;
  const float* ar = agg2 + (size_t)rowc * HID_F;

  // A: K = g*16 + h*8 + w ; valid iff K < 16 (g==0), rest zero-padded
  float4 q0 = *(const float4*)(ar + h * 8);
  float4 q1 = *(const float4*)(ar + h * 8 + 4);
  v16h a;
  a[0] = (_Float16)q0.x; a[1] = (_Float16)q0.y;
  a[2] = (_Float16)q0.z; a[3] = (_Float16)q0.w;
  a[4] = (_Float16)q1.x; a[5] = (_Float16)q1.y;
  a[6] = (_Float16)q1.z; a[7] = (_Float16)q1.w;
  #pragma unroll
  for (int e = 8; e < 16; ++e) a[e] = (_Float16)0.0f;

  #pragma unroll
  for (int t = 0; t < 4; ++t) {              // 4 column tiles of the 16x64 output
    v16h b = W2f[t * 32 + lane];
    v8f c = {0.f, 0.f, 0.f, 0.f, 0.f, 0.f, 0.f, 0.f};
    c = __builtin_amdgcn_wmma_f32_16x16x32_f16(false, a, false, b, (short)0, c,
                                               false, false);
    #pragma unroll
    for (int i = 0; i < 8; ++i) {
      int r = wave * 16 + i + h * 8;
      if (r < N)
        out[(size_t)r * OUT_F + t * 16 + m] = c[i] * norm_dst[r] + b2[t * 16 + m];
    }
  }
}

extern "C" void kernel_launch(void* const* d_in, const int* in_sizes, int n_in,
                              void* d_out, int out_size, void* d_ws, size_t ws_size,
                              hipStream_t stream) {
  const float* x  = (const float*)d_in[0];
  const int*   src = (const int*)d_in[1];
  const int*   dst = (const int*)d_in[2];
  const float* W1 = (const float*)d_in[3];
  const float* b1 = (const float*)d_in[4];
  const float* W2 = (const float*)d_in[5];
  const float* b2 = (const float*)d_in[6];
  float* out = (float*)d_out;

  const int N = in_sizes[0] / IN_F;
  const int E = in_sizes[1];

  // Workspace layout (4-byte words):
  // [W1f 1024][W2f 1024][deg_src N][deg_dst N][agg1 16N][agg2 16N]
  // [norm_src N][norm_dst N][tmp1/feat2 16N]
  float* ws = (float*)d_ws;
  _Float16* W1f = (_Float16*)ws;                           // 2048 halves (32B-aligned)
  _Float16* W2f = W1f + 2048;                              // 2048 halves
  unsigned* deg_src = (unsigned*)(ws + 2048);              // N
  unsigned* deg_dst = deg_src + (size_t)N;                 // N
  float* agg1     = ws + 2048 + 2 * (size_t)N;             // 16N
  float* agg2     = agg1 + 16 * (size_t)N;                 // 16N
  float* norm_src = agg2 + 16 * (size_t)N;                 // N
  float* norm_dst = norm_src + (size_t)N;                  // N
  float* tmp1     = norm_dst + (size_t)N;                  // 16N (reused as feat2)

  long zero_words = 34L * N;                               // deg_src..agg2
  k_zero<<<(int)((zero_words + 255) / 256), 256, 0, stream>>>(ws + 2048, zero_words);

  k_prep<<<16, 256, 0, stream>>>(W1, W2, W1f, W2f);
  k_degree<<<(E + 255) / 256, 256, 0, stream>>>(src, dst, deg_src, deg_dst, E);
  k_norm<<<(N + 255) / 256, 256, 0, stream>>>(deg_src, deg_dst, norm_src, norm_dst, N);

  int tiles   = (N + 15) / 16;
  int gblocks = (tiles + 7) / 8;                           // 256 thr = 8 waves/block
  k_gemm1<<<gblocks, 256, 0, stream>>>(x, (const v16h*)W1f, norm_src, tmp1, N);

  long EF = 16L * E;
  long NF = 16L * N;
  k_scatter<<<(int)((EF + 255) / 256), 256, 0, stream>>>(src, dst, tmp1, agg1, EF);
  k_post1<<<(int)((NF + 255) / 256), 256, 0, stream>>>(agg1, norm_dst, norm_src, b1,
                                                       tmp1, NF);
  k_scatter<<<(int)((EF + 255) / 256), 256, 0, stream>>>(src, dst, tmp1, agg2, EF);
  k_gemm2<<<gblocks, 256, 0, stream>>>(agg2, (const v16h*)W2f, norm_dst, b2, out, N);
}
